// QLinearLayer_60507499266261
// MI455X (gfx1250) — matrix operand order repaired and checked by reading, not compiled, not run on hardware
//
#include <hip/hip_runtime.h>
#include <hip/hip_bf16.h>
#include <stdint.h>

typedef __attribute__((ext_vector_type(16))) _Float16 v16h;
typedef __attribute__((ext_vector_type(8)))  float    v8f;
typedef __attribute__((ext_vector_type(4)))  float    v4f;
typedef __attribute__((ext_vector_type(4)))  int      v4i;

// Address-space-typed pointers for the gfx1250 async-copy builtin
// (param 1 diag-printed as "int __vector(4) __device__ *" => int4 addrspace(1)*)
typedef __attribute__((address_space(1))) v4i* gv4i_p;   // global int4*
typedef __attribute__((address_space(3))) v4i* lv4i_p;   // LDS int4*

#define TM 128        // workgroup tile M
#define TN 128        // workgroup tile N
#define TK 32         // k-slice per stage (one WMMA K)
#define LDT (TK + 8)  // padded LDS row stride (halfs): 80B rows -> conflict-free b128 reads

#if __has_builtin(__builtin_amdgcn_global_load_async_to_lds_b128)
#define HAVE_ASYNC 1
#else
#define HAVE_ASYNC 0
#endif

#define NBUF (HAVE_ASYNC ? 2 : 1)

__device__ __forceinline__ void wait_async0()
{
#if __has_builtin(__builtin_amdgcn_s_wait_asynccnt)
    __builtin_amdgcn_s_wait_asynccnt(0);
#else
    asm volatile("s_wait_asynccnt 0x0" ::: "memory");
#endif
}

// ---------------------------------------------------------------------------
// Pass 1: dequantize fp4-grid values (stored as f32) * per-16 block scale -> f16
// Streams 268MB once (non-temporal reads), leaves 96MB f16 hot for the GEMM.
// ---------------------------------------------------------------------------
__global__ __launch_bounds__(256)
void dequant_to_f16(const float* __restrict__ q, const float* __restrict__ s,
                    _Float16* __restrict__ outh, long long nblocks)
{
    long long t = (long long)blockIdx.x * blockDim.x + threadIdx.x;
    if (t >= nblocks) return;
    const float sc = __builtin_nontemporal_load(s + t);
    const v4f* src = (const v4f*)(q + t * 16);
    uint4* dst = (uint4*)(outh + t * 16);
    union { _Float16 h[8]; uint4 u; } pk;
    #pragma unroll
    for (int i = 0; i < 2; ++i) {
        v4f a = __builtin_nontemporal_load(src + i * 2 + 0);
        v4f b = __builtin_nontemporal_load(src + i * 2 + 1);
        pk.h[0] = (_Float16)(a.x * sc); pk.h[1] = (_Float16)(a.y * sc);
        pk.h[2] = (_Float16)(a.z * sc); pk.h[3] = (_Float16)(a.w * sc);
        pk.h[4] = (_Float16)(b.x * sc); pk.h[5] = (_Float16)(b.y * sc);
        pk.h[6] = (_Float16)(b.z * sc); pk.h[7] = (_Float16)(b.w * sc);
        dst[i] = pk.u;     // regular store: keep f16 operands resident in L2
    }
}

// ---------------------------------------------------------------------------
// Fragment loads (ISA 7.12.2 16-bit layouts), 2x ds_load_b128 per lane each.
// ---------------------------------------------------------------------------
__device__ __forceinline__ v16h load_frag_a(const _Float16* tile, int lane)
{
    const int r = lane & 15, h = lane >> 4;
    const _Float16* p = tile + r * LDT + h * 8;
    union { v16h v; uint4 q[2]; } u;
    u.q[0] = *(const uint4*)(p);
    u.q[1] = *(const uint4*)(p + 16);
    return u.v;
}

__device__ __forceinline__ v16h load_frag_b(const _Float16* tile, int lane)
{
    const int r = lane & 15, h = lane >> 4;
    const _Float16* p = tile + r * LDT + h * 8;
    union { v16h v; uint4 q[2]; } u;
    u.q[0] = *(const uint4*)(p);
    u.q[1] = *(const uint4*)(p + 16);
    return u.v;
}

// 4x2 WMMA tiles per wave on the staged k-slice
__device__ __forceinline__ void compute_tile(const _Float16* Xt, const _Float16* Wt,
                                             int lane, int wm, int wn, v8f (&acc)[4][2])
{
    v16h a[4], b[2];
    #pragma unroll
    for (int i = 0; i < 4; ++i) a[i] = load_frag_a(Xt + (wm + i * 16) * LDT, lane);
    #pragma unroll
    for (int j = 0; j < 2; ++j) b[j] = load_frag_b(Wt + (wn + j * 16) * LDT, lane);
    #pragma unroll
    for (int i = 0; i < 4; ++i)
        #pragma unroll
        for (int j = 0; j < 2; ++j)
            acc[i][j] = __builtin_amdgcn_wmma_f32_16x16x32_f16(
                false, a[i], false, b[j], (short)0, acc[i][j], false, false);
}

#if HAVE_ASYNC
// Async-copy one 128x32 f16 tile (global row-major, ld=K) into a padded LDS tile.
// 2 x GLOBAL_LOAD_ASYNC_TO_LDS_B128 per thread (512 16B chunks / 256 threads).
__device__ __forceinline__ void stage_async(const _Float16* src, int K,
                                            _Float16* dst, int tid)
{
    #pragma unroll
    for (int i = 0; i < 2; ++i) {
        const int c = tid + i * 256;
        const int row = c >> 2, col = (c & 3) * 8;
        __builtin_amdgcn_global_load_async_to_lds_b128(
            (gv4i_p)(uintptr_t)(src + (size_t)row * K + col),
            (lv4i_p)(unsigned)(uintptr_t)(dst + row * LDT + col),
            0, 0);
    }
}
#else
__device__ __forceinline__ void fetch_regs(const _Float16* src, int K, int k0,
                                           uint4 (&r)[2], int tid)
{
    #pragma unroll
    for (int i = 0; i < 2; ++i) {
        const int c = tid + i * 256;
        const int row = c >> 2, col = (c & 3) * 8;
        r[i] = *(const uint4*)(src + (size_t)row * K + k0 + col);
    }
}
__device__ __forceinline__ void store_regs(const uint4 (&r)[2], _Float16* dst, int tid)
{
    #pragma unroll
    for (int i = 0; i < 2; ++i) {
        const int c = tid + i * 256;
        const int row = c >> 2, col = (c & 3) * 8;
        *(uint4*)(dst + row * LDT + col) = r[i];
    }
}
#endif

// ---------------------------------------------------------------------------
// Pass 2: GEMM  out[M,N] = Xd[M,K] * Wd[N,K]^T * gscale + bias
// 8 wave32/WG, 128x128 tile, double-buffered async LDS pipeline.
// ---------------------------------------------------------------------------
__global__ __launch_bounds__(256)
void nvfp4_gemm_wmma_f16(const _Float16* __restrict__ Xh, const _Float16* __restrict__ Wh,
                         const float* __restrict__ sgx, const float* __restrict__ sgw,
                         const float* __restrict__ bias, float* __restrict__ out,
                         int M, int N, int K)
{
    __shared__ __align__(16) _Float16 Xs[NBUF][TM][LDT];
    __shared__ __align__(16) _Float16 Ws[NBUF][TN][LDT];

    const int tid  = threadIdx.x;
    const int lane = tid & 31;
    const int wave = tid >> 5;
    const int wm   = (wave & 1) * 64;
    const int wn   = (wave >> 1) * 32;
    const int m0   = blockIdx.y * TM;
    const int n0   = blockIdx.x * TN;

    const _Float16* xsrc = Xh + (size_t)m0 * K;
    const _Float16* wsrc = Wh + (size_t)n0 * K;
    const int nt = K / TK;

    v8f acc[4][2] = {};

#if HAVE_ASYNC
    stage_async(xsrc, K, &Xs[0][0][0], tid);
    stage_async(wsrc, K, &Ws[0][0][0], tid);
    for (int t = 0; t < nt; ++t) {
        const int cur = t & 1;
        wait_async0();                 // my tile-t async copies have landed
        __syncthreads();               // everyone's landed; tile t-1 reads all done
        if (t + 1 < nt) {              // overlap: stream tile t+1 behind 8 WMMAs
            stage_async(xsrc + (size_t)(t + 1) * TK, K, &Xs[cur ^ 1][0][0], tid);
            stage_async(wsrc + (size_t)(t + 1) * TK, K, &Ws[cur ^ 1][0][0], tid);
        }
        compute_tile(&Xs[cur][0][0], &Ws[cur][0][0], lane, wm, wn, acc);
    }
#else
    // Fallback: register-prefetch pipeline (loads for t+1 in flight during compute of t)
    uint4 rx[2], rw[2];
    fetch_regs(xsrc, K, 0, rx, tid);
    fetch_regs(wsrc, K, 0, rw, tid);
    for (int t = 0; t < nt; ++t) {
        __syncthreads();               // all reads of previous tile finished
        store_regs(rx, &Xs[0][0][0], tid);
        store_regs(rw, &Ws[0][0][0], tid);
        if (t + 1 < nt) {
            fetch_regs(xsrc, K, (t + 1) * TK, rx, tid);
            fetch_regs(wsrc, K, (t + 1) * TK, rw, tid);
        }
        __syncthreads();               // LDS tile ready (implied s_wait_dscnt)
        compute_tile(&Xs[0][0][0], &Ws[0][0][0], lane, wm, wn, acc);
    }
#endif

    // Epilogue: global scale + bias; C/D layout: lane -> N=l&15, row = r + 8*(l>>4)
    const float gscale = sgx[0] * sgw[0];
    const int cn   = lane & 15;
    const int rsel = (lane >> 4) * 8;
    #pragma unroll
    for (int i = 0; i < 4; ++i) {
        #pragma unroll
        for (int j = 0; j < 2; ++j) {
            const int gn = n0 + wn + j * 16 + cn;
            const float bv = bias[gn];
            #pragma unroll
            for (int r = 0; r < 8; ++r) {
                const int gm = m0 + wm + i * 16 + r + rsel;
                __builtin_nontemporal_store(acc[i][j][r] * gscale + bv,
                                            out + (size_t)gm * N + gn);
            }
        }
    }
}

// ---------------------------------------------------------------------------
// Fused fallback (only if workspace too small): dequantize tiles in the GEMM.
// ---------------------------------------------------------------------------
__global__ __launch_bounds__(256)
void nvfp4_gemm_wmma_fused(const float* __restrict__ qx, const float* __restrict__ sx,
                           const float* __restrict__ qw, const float* __restrict__ sw,
                           const float* __restrict__ sgx, const float* __restrict__ sgw,
                           const float* __restrict__ bias, float* __restrict__ out,
                           int M, int N, int K)
{
    __shared__ __align__(16) _Float16 Xs[TM][LDT];
    __shared__ __align__(16) _Float16 Ws[TN][LDT];

    const int tid  = threadIdx.x;
    const int lane = tid & 31;
    const int wave = tid >> 5;
    const int wm   = (wave & 1) * 64;
    const int wn   = (wave >> 1) * 32;
    const int m0   = blockIdx.y * TM;
    const int n0   = blockIdx.x * TN;
    const int Bk   = K >> 4;

    v8f acc[4][2] = {};

    for (int k0 = 0; k0 < K; k0 += TK) {
        #pragma unroll
        for (int u = 0; u < 4; ++u) {
            const int c = tid + u * 256;            // 1024 4-float chunks
            const int row = c >> 3, col = (c & 7) * 4;
            {
                const int gm = m0 + row, gk = k0 + col;
                v4f v = *(const v4f*)(qx + (size_t)gm * K + gk);
                float sc = sx[(size_t)gm * Bk + (gk >> 4)];
                union { _Float16 h[4]; uint2 uu; } pk;
                pk.h[0] = (_Float16)(v.x * sc); pk.h[1] = (_Float16)(v.y * sc);
                pk.h[2] = (_Float16)(v.z * sc); pk.h[3] = (_Float16)(v.w * sc);
                *(uint2*)&Xs[row][col] = pk.uu;
            }
            {
                const int gn = n0 + row, gk = k0 + col;
                v4f v = *(const v4f*)(qw + (size_t)gn * K + gk);
                float sc = sw[(size_t)gn * Bk + (gk >> 4)];
                union { _Float16 h[4]; uint2 uu; } pk;
                pk.h[0] = (_Float16)(v.x * sc); pk.h[1] = (_Float16)(v.y * sc);
                pk.h[2] = (_Float16)(v.z * sc); pk.h[3] = (_Float16)(v.w * sc);
                *(uint2*)&Ws[row][col] = pk.uu;
            }
        }
        __syncthreads();
        compute_tile(&Xs[0][0], &Ws[0][0], lane, wm, wn, acc);
        __syncthreads();
    }

    const float gscale = sgx[0] * sgw[0];
    const int cn   = lane & 15;
    const int rsel = (lane >> 4) * 8;
    #pragma unroll
    for (int i = 0; i < 4; ++i) {
        #pragma unroll
        for (int j = 0; j < 2; ++j) {
            const int gn = n0 + wn + j * 16 + cn;
            const float bv = bias[gn];
            #pragma unroll
            for (int r = 0; r < 8; ++r) {
                const int gm = m0 + wm + i * 16 + r + rsel;
                __builtin_nontemporal_store(acc[i][j][r] * gscale + bv,
                                            out + (size_t)gm * N + gn);
            }
        }
    }
}

// ---------------------------------------------------------------------------
extern "C" void kernel_launch(void* const* d_in, const int* in_sizes, int n_in,
                              void* d_out, int out_size, void* d_ws, size_t ws_size,
                              hipStream_t stream)
{
    const float* qx  = (const float*)d_in[0];   // [M,K]
    const float* sx  = (const float*)d_in[1];   // [M,K/16]
    const float* sgx = (const float*)d_in[2];   // scalar
    const float* qw  = (const float*)d_in[3];   // [N,K]
    const float* sw  = (const float*)d_in[4];   // [N,K/16]
    const float* sgw = (const float*)d_in[5];   // scalar
    const float* bias= (const float*)d_in[6];   // [N]
    float* out = (float*)d_out;

    const int N = in_sizes[6];
    const int K = in_sizes[3] / N;
    const int M = in_sizes[0] / K;
    const int Bk = K / 16;

    dim3 grid(N / TN, M / TM), block(256);

    const size_t need = ((size_t)M * K + (size_t)N * K) * sizeof(_Float16);
    if (ws_size >= need) {
        _Float16* Xh = (_Float16*)d_ws;
        _Float16* Wh = Xh + (size_t)M * K;
        long long nbx = (long long)M * Bk;
        long long nbw = (long long)N * Bk;
        dequant_to_f16<<<(unsigned)((nbx + 255) / 256), 256, 0, stream>>>(qx, sx, Xh, nbx);
        dequant_to_f16<<<(unsigned)((nbw + 255) / 256), 256, 0, stream>>>(qw, sw, Wh, nbw);
        nvfp4_gemm_wmma_f16<<<grid, block, 0, stream>>>(Xh, Wh, sgx, sgw, bias, out, M, N, K);
    } else {
        nvfp4_gemm_wmma_fused<<<grid, block, 0, stream>>>(qx, sx, qw, sw, sgx, sgw,
                                                          bias, out, M, N, K);
    }
}
